// InteractionNetworkLayer_68461778698611
// MI455X (gfx1250) — matrix operand order repaired and checked by reading, not compiled, not run on hardware
//
#include <hip/hip_runtime.h>

// ---------------------------------------------------------------------------
// CDNA5 (gfx1250) Interaction-Network layer, register-blocked WMMA version.
//   edge MLP : x[64x640] @ eW1[640x256] -> relu -> @ eW2[256x128]
//   scatter  : atomic f32 add into agg[B][N][128]   (L2-resident, 16 MB)
//   node MLP : x[64x384] @ nW1[384x256] -> relu -> @ nW2[256x256]
//   residual : nodes_out = nodes + ..., edges_out = edges + ...
// - 4 v_wmma_f32_16x16x32_bf16 per weight-fragment fetch (M-blocking).
// - LDS row strides padded to 4 (mod 64) dwords: conflict-free ds_load_b128.
// - Stage-0 gathers use float4 loads + packed 4x bf16 LDS stores.
// ---------------------------------------------------------------------------

typedef __bf16 bf16_t;
typedef __attribute__((ext_vector_type(16))) __bf16 v16bf;
typedef __attribute__((ext_vector_type(8)))  __bf16 v8bf;
typedef __attribute__((ext_vector_type(4)))  __bf16 v4bf;
typedef __attribute__((ext_vector_type(8)))  float  v8f;
typedef __attribute__((ext_vector_type(4)))  float  v4f;

#define B_    2
#define NN_   16384
#define EE_   131072
#define DN_   256
#define DE_   128
#define HH_   256
#define KE_   640   // 2*DN + DE
#define KN_   384   // DE + DN
#define ROWS_ 64    // rows (edges/nodes) per workgroup

// Padded LDS row strides (bf16 units). stride/2 mod 64 dwords == 4 =>
// lanes 0..15 (different rows, same col) hit disjoint 4-bank groups.
#define XE_ST 648   // edge x rows   (324 dw % 64 == 4)
#define HS_ST 264   // hidden rows   (132 dw % 64 == 4)
#define XN_ST 392   // node x rows   (196 dw % 64 == 4)

// A-fragment for v_wmma_f32_16x16x32_bf16 (contiguous k order, same as the B-frag):
// lane L: m = L&15, hi = L>>4.  a[e] = x[m][kk + hi*16 + e]
static __device__ inline v16bf load_a16(const bf16_t* rowk, int hi) {
  v8bf lo = *(const v8bf*)(rowk + hi * 16);
  v8bf hh = *(const v8bf*)(rowk + hi * 16 + 8);
  v16bf a;
#pragma unroll
  for (int i = 0; i < 8; ++i) { a[i] = lo[i]; a[8 + i] = hh[i]; }
  return a;
}

static __device__ inline v8f wmma_bf16(v16bf a, v16bf b, v8f c) {
  return __builtin_amdgcn_wmma_f32_16x16x32_bf16(false, a, false, b,
                                                 (short)0, c, false, false);
}

static __device__ inline v4bf cvt4(v4f v) {
  v4bf o;
#pragma unroll
  for (int j = 0; j < 4; ++j) o[j] = (bf16_t)v[j];
  return o;
}

// ---------------------------------------------------------------------------
__global__ void zero_kernel(float* __restrict__ p, int n) {
  int i = blockIdx.x * blockDim.x + threadIdx.x;
  if (i < n) p[i] = 0.0f;
}

// f32 W[K][N] -> bf16 transposed WT[N][K]; B-frag reads 16 contiguous bf16:
// b[e] = W[hi*16+e][n] = WT[n][hi*16+e].
__global__ void wprep_kernel(const float* __restrict__ W, bf16_t* __restrict__ WT,
                             int K, int N) {
  int i = blockIdx.x * blockDim.x + threadIdx.x;
  if (i < K * N) {
    int n = i / K, k = i - n * K;
    WT[i] = (bf16_t)W[(size_t)k * N + n];
  }
}

// ---------------------------------------------------------------------------
// Edge MLP: 64 edges per workgroup, 16 waves.
__global__ __launch_bounds__(512)
void edge_mlp_kernel(const float* __restrict__ nodes,
                     const float* __restrict__ edges,
                     const int*   __restrict__ senders,
                     const int*   __restrict__ receivers,
                     const bf16_t* __restrict__ WT1,  // [256][640]
                     const float* __restrict__ b1,    // [256]
                     const bf16_t* __restrict__ WT2,  // [128][256]
                     const float* __restrict__ b2,    // [128]
                     float* __restrict__ out_edges,   // [B*E*De]
                     float* __restrict__ agg)         // [B*N*De]
{
  extern __shared__ __align__(16) char smem[];
  bf16_t* xs = (bf16_t*)smem;                                    // [64][XE_ST]
  bf16_t* hs = (bf16_t*)(smem + (size_t)ROWS_ * XE_ST * 2);      // [64][HS_ST]
  int* ridx  = (int*)(smem + (size_t)ROWS_ * (XE_ST + HS_ST) * 2);
  int* sidx  = ridx + ROWS_;

  const int  tid = threadIdx.x;
  const long f0  = (long)blockIdx.x * ROWS_;
  const int  b   = (int)(f0 / EE_);
  const int  e0  = (int)(f0 - (long)b * EE_);

  if (tid < ROWS_) {
    ridx[tid] = receivers[(size_t)b * EE_ + e0 + tid];
    sidx[tid] = senders  [(size_t)b * EE_ + e0 + tid];
  }
  __syncthreads();

  // Stage 0: gather recv||send||edge features -> LDS bf16, float4 granular.
  {
    const int QK = KE_ / 4;  // 160 float4 per row
    for (int i = tid; i < ROWS_ * QK; i += 512) {
      int row = i / QK, c = (i - row * QK) * 4;
      v4f v;
      if (c < DN_)
        v = *(const v4f*)(nodes + ((size_t)b * NN_ + ridx[row]) * DN_ + c);
      else if (c < 2 * DN_)
        v = *(const v4f*)(nodes + ((size_t)b * NN_ + sidx[row]) * DN_ + (c - DN_));
      else
        v = *(const v4f*)(edges + ((size_t)b * EE_ + e0 + row) * DE_ + (c - 2 * DN_));
      *(v4bf*)(xs + row * XE_ST + c) = cvt4(v);
    }
  }
  __syncthreads();

  const int lane = tid & 31, w = tid >> 5;
  const int q = lane & 15, hi = lane >> 4;

  // Stage 1: h = relu(x @ eW1 + b1). Wave w: H-cols [16w,16w+16), 4 M-tiles.
  {
    v8f acc0 = {}, acc1 = {}, acc2 = {}, acc3 = {};
    const bf16_t* wrow = WT1 + (size_t)(w * 16 + q) * KE_;
    const bf16_t* xrow = xs + q * XE_ST;
#pragma unroll 2
    for (int kk = 0; kk < KE_; kk += 32) {
      v16bf bm = *(const v16bf*)(wrow + kk + hi * 16);     // shared B frag
      v16bf a0 = load_a16(xrow + 0 * 16 * XE_ST + kk, hi);
      v16bf a1 = load_a16(xrow + 1 * 16 * XE_ST + kk, hi);
      v16bf a2 = load_a16(xrow + 2 * 16 * XE_ST + kk, hi);
      v16bf a3 = load_a16(xrow + 3 * 16 * XE_ST + kk, hi);
      acc0 = wmma_bf16(a0, bm, acc0);
      acc1 = wmma_bf16(a1, bm, acc1);
      acc2 = wmma_bf16(a2, bm, acc2);
      acc3 = wmma_bf16(a3, bm, acc3);
    }
    const float bias = b1[w * 16 + q];
    v8f av[4] = {acc0, acc1, acc2, acc3};
#pragma unroll
    for (int mt = 0; mt < 4; ++mt)
#pragma unroll
      for (int r = 0; r < 8; ++r) {        // C layout: M = r + 8*hi, N = q
        float v = av[mt][r] + bias;
        hs[(mt * 16 + r + 8 * hi) * HS_ST + w * 16 + q] = (bf16_t)(v > 0.f ? v : 0.f);
      }
  }
  __syncthreads();

  // Stage 2: out = h @ eW2 + b2. 8 N-tiles x 4 M-tiles over 16 waves
  // -> wave w: N-tile (w&7), M-tiles {2*(w>>3), 2*(w>>3)+1}, shared B frag.
  {
    const int nt = w & 7, mtb = (w >> 3) * 2;
    v8f acc[2]; acc[0] = v8f{}; acc[1] = v8f{};
    const bf16_t* wrow = WT2 + (size_t)(nt * 16 + q) * HH_;
    const bf16_t* hrow = hs + (mtb * 16 + q) * HS_ST;
#pragma unroll
    for (int kk = 0; kk < HH_; kk += 32) {
      v16bf bm = *(const v16bf*)(wrow + kk + hi * 16);
      v16bf a0 = load_a16(hrow + kk, hi);
      v16bf a1 = load_a16(hrow + 16 * HS_ST + kk, hi);
      acc[0] = wmma_bf16(a0, bm, acc[0]);
      acc[1] = wmma_bf16(a1, bm, acc[1]);
    }
    const int n = nt * 16 + q;
    const float bias = b2[n];
#pragma unroll
    for (int mt = 0; mt < 2; ++mt)
#pragma unroll
      for (int r = 0; r < 8; ++r) {
        const int row = (mtb + mt) * 16 + r + 8 * hi;
        const float ev = acc[mt][r] + bias;                 // edges_new
        const size_t eoff = ((size_t)b * EE_ + e0 + row) * DE_ + n;
        out_edges[eoff] = edges[eoff] + ev;                 // residual out
        atomicAdd(&agg[((size_t)b * NN_ + ridx[row]) * DE_ + n], ev); // segsum
      }
  }
}

// ---------------------------------------------------------------------------
// Node MLP: 64 nodes per workgroup, 16 waves.
__global__ __launch_bounds__(512)
void node_mlp_kernel(const float* __restrict__ nodes,
                     const float* __restrict__ agg,
                     const bf16_t* __restrict__ WT1,  // [256][384]
                     const float* __restrict__ b1,
                     const bf16_t* __restrict__ WT2,  // [256][256]
                     const float* __restrict__ b2,
                     float* __restrict__ out_nodes)   // [B*N*Dn]
{
  extern __shared__ __align__(16) char smem[];
  bf16_t* xs = (bf16_t*)smem;                               // [64][XN_ST]
  bf16_t* hs = (bf16_t*)(smem + (size_t)ROWS_ * XN_ST * 2); // [64][HS_ST]

  const int  tid = threadIdx.x;
  const long f0  = (long)blockIdx.x * ROWS_;
  const int  b   = (int)(f0 / NN_);
  const int  n0  = (int)(f0 - (long)b * NN_);

  // Stage 0: concat(agg, nodes) -> LDS bf16, float4 granular.
  {
    const int QK = KN_ / 4;  // 96
    for (int i = tid; i < ROWS_ * QK; i += 512) {
      int row = i / QK, c = (i - row * QK) * 4;
      v4f v;
      if (c < DE_)
        v = *(const v4f*)(agg + ((size_t)b * NN_ + n0 + row) * DE_ + c);
      else
        v = *(const v4f*)(nodes + ((size_t)b * NN_ + n0 + row) * DN_ + (c - DE_));
      *(v4bf*)(xs + row * XN_ST + c) = cvt4(v);
    }
  }
  __syncthreads();

  const int lane = tid & 31, w = tid >> 5;
  const int q = lane & 15, hi = lane >> 4;

  // Stage 1: h = relu(x @ nW1 + b1). Wave w: cols [16w,16w+16), 4 M-tiles.
  {
    v8f acc0 = {}, acc1 = {}, acc2 = {}, acc3 = {};
    const bf16_t* wrow = WT1 + (size_t)(w * 16 + q) * KN_;
    const bf16_t* xrow = xs + q * XN_ST;
#pragma unroll 2
    for (int kk = 0; kk < KN_; kk += 32) {
      v16bf bm = *(const v16bf*)(wrow + kk + hi * 16);
      v16bf a0 = load_a16(xrow + 0 * 16 * XN_ST + kk, hi);
      v16bf a1 = load_a16(xrow + 1 * 16 * XN_ST + kk, hi);
      v16bf a2 = load_a16(xrow + 2 * 16 * XN_ST + kk, hi);
      v16bf a3 = load_a16(xrow + 3 * 16 * XN_ST + kk, hi);
      acc0 = wmma_bf16(a0, bm, acc0);
      acc1 = wmma_bf16(a1, bm, acc1);
      acc2 = wmma_bf16(a2, bm, acc2);
      acc3 = wmma_bf16(a3, bm, acc3);
    }
    const float bias = b1[w * 16 + q];
    v8f av[4] = {acc0, acc1, acc2, acc3};
#pragma unroll
    for (int mt = 0; mt < 4; ++mt)
#pragma unroll
      for (int r = 0; r < 8; ++r) {
        float v = av[mt][r] + bias;
        hs[(mt * 16 + r + 8 * hi) * HS_ST + w * 16 + q] = (bf16_t)(v > 0.f ? v : 0.f);
      }
  }
  __syncthreads();

  // Stage 2: nodes_out = nodes + (h @ nW2 + b2). Wave w: N-tile w, 4 M-tiles.
  {
    v8f acc0 = {}, acc1 = {}, acc2 = {}, acc3 = {};
    const bf16_t* wrow = WT2 + (size_t)(w * 16 + q) * HH_;
    const bf16_t* hrow = hs + q * HS_ST;
#pragma unroll 2
    for (int kk = 0; kk < HH_; kk += 32) {
      v16bf bm = *(const v16bf*)(wrow + kk + hi * 16);
      v16bf a0 = load_a16(hrow + 0 * 16 * HS_ST + kk, hi);
      v16bf a1 = load_a16(hrow + 1 * 16 * HS_ST + kk, hi);
      v16bf a2 = load_a16(hrow + 2 * 16 * HS_ST + kk, hi);
      v16bf a3 = load_a16(hrow + 3 * 16 * HS_ST + kk, hi);
      acc0 = wmma_bf16(a0, bm, acc0);
      acc1 = wmma_bf16(a1, bm, acc1);
      acc2 = wmma_bf16(a2, bm, acc2);
      acc3 = wmma_bf16(a3, bm, acc3);
    }
    const int n = w * 16 + q;
    const float bias = b2[n];
    v8f av[4] = {acc0, acc1, acc2, acc3};
#pragma unroll
    for (int mt = 0; mt < 4; ++mt)
#pragma unroll
      for (int r = 0; r < 8; ++r) {
        const int row = mt * 16 + r + 8 * hi;
        const size_t off = ((size_t)b * NN_ + n0 + row) * DN_ + n;
        out_nodes[off] = nodes[off] + av[mt][r] + bias;
      }
  }
}

// ---------------------------------------------------------------------------
extern "C" void kernel_launch(void* const* d_in, const int* in_sizes, int n_in,
                              void* d_out, int out_size, void* d_ws, size_t ws_size,
                              hipStream_t stream) {
  (void)in_sizes; (void)n_in; (void)out_size; (void)ws_size;

  const float* nodes     = (const float*)d_in[0];
  const float* edges     = (const float*)d_in[1];
  const int*   senders   = (const int*)d_in[2];
  const int*   receivers = (const int*)d_in[3];
  const float* eW1 = (const float*)d_in[4];
  const float* eb1 = (const float*)d_in[5];
  const float* eW2 = (const float*)d_in[6];
  const float* eb2 = (const float*)d_in[7];
  const float* nW1 = (const float*)d_in[8];
  const float* nb1 = (const float*)d_in[9];
  const float* nW2 = (const float*)d_in[10];
  const float* nb2 = (const float*)d_in[11];

  // Output: nodes_out [B*N*Dn] first, then edges_out [B*E*De]
  float* out_nodes = (float*)d_out;
  float* out_edges = (float*)d_out + (size_t)B_ * NN_ * DN_;

  // Workspace: agg (16 MB, f32) then transposed bf16 weights (~0.7 MB)
  char* ws = (char*)d_ws;
  float* agg = (float*)ws;
  size_t off = (size_t)B_ * NN_ * DE_ * sizeof(float);
  bf16_t* WT_e1 = (bf16_t*)(ws + off); off += (size_t)KE_ * HH_ * sizeof(bf16_t);
  bf16_t* WT_e2 = (bf16_t*)(ws + off); off += (size_t)HH_ * DE_ * sizeof(bf16_t);
  bf16_t* WT_n1 = (bf16_t*)(ws + off); off += (size_t)KN_ * HH_ * sizeof(bf16_t);
  bf16_t* WT_n2 = (bf16_t*)(ws + off);

  const int aggN = B_ * NN_ * DE_;
  zero_kernel<<<(aggN + 255) / 256, 256, 0, stream>>>(agg, aggN);
  wprep_kernel<<<(KE_ * HH_ + 255) / 256, 256, 0, stream>>>(eW1, WT_e1, KE_, HH_);
  wprep_kernel<<<(HH_ * DE_ + 255) / 256, 256, 0, stream>>>(eW2, WT_e2, HH_, DE_);
  wprep_kernel<<<(KN_ * HH_ + 255) / 256, 256, 0, stream>>>(nW1, WT_n1, KN_, HH_);
  wprep_kernel<<<(HH_ * DN_ + 255) / 256, 256, 0, stream>>>(nW2, WT_n2, HH_, DN_);

  const size_t edge_lds = (size_t)ROWS_ * (XE_ST + HS_ST) * sizeof(bf16_t)
                        + 2 * ROWS_ * sizeof(int);                       // ~117 KB
  const size_t node_lds = (size_t)ROWS_ * (XN_ST + HS_ST) * sizeof(bf16_t); // ~82 KB

  edge_mlp_kernel<<<(B_ * EE_) / ROWS_, 512, edge_lds, stream>>>(
      nodes, edges, senders, receivers, WT_e1, eb1, WT_e2, eb2, out_edges, agg);
  node_mlp_kernel<<<(B_ * NN_) / ROWS_, 512, node_lds, stream>>>(
      nodes, agg, WT_n1, nb1, WT_n2, nb2, out_nodes);
}
